// SelfAttention_90323162235073
// MI455X (gfx1250) — matrix operand order repaired and checked by reading, not compiled, not run on hardware
//
#include <hip/hip_runtime.h>

#define B_ 8
#define N_ 2048
#define D_ 512

typedef __bf16 bf16;
typedef __attribute__((ext_vector_type(8)))  __bf16 v8bf;
typedef __attribute__((ext_vector_type(16))) __bf16 v16bf;
typedef __attribute__((ext_vector_type(8)))  float  v8f;

static __device__ __forceinline__ v8bf ld8(const bf16* p) {
    return *(const v8bf*)p;
}
static __device__ __forceinline__ v16bf cat16(v8bf lo, v8bf hi) {
    return __builtin_shufflevector(lo, hi, 0,1,2,3,4,5,6,7,8,9,10,11,12,13,14,15);
}

// CDNA5 async copy: one b128 per lane, global -> LDS, tracked by ASYNCcnt.
static __device__ __forceinline__ void async_copy16(unsigned lds_off, const bf16* gsrc) {
    asm volatile("global_load_async_to_lds_b128 %0, %1, off"
                 :: "v"(lds_off), "v"(gsrc) : "memory");
}
static __device__ __forceinline__ void wait_async0() {
    asm volatile("s_wait_asynccnt 0x0" ::: "memory");
}

// ---------------------------------------------------------------------------
// Kernel 0: one-time conversions (memory-bound, ~50 MB total traffic).
//   blocks [0, 4096):    x (f32) -> Xb (bf16), 8 elems/thread, vectorized
//   blocks [4096, 7168): W[3] (f32, [k][n]) -> Wt[3] (bf16, [n][k]) transposed
// ---------------------------------------------------------------------------
__global__ __launch_bounds__(256) void prep(
    const float* __restrict__ x,
    const float* __restrict__ Wq, const float* __restrict__ Wk,
    const float* __restrict__ Wv,
    bf16* __restrict__ Xb, bf16* __restrict__ Wt)
{
    const int blk = blockIdx.x;
    if (blk < 4096) {
        const size_t e = ((size_t)blk * 256 + threadIdx.x) * 8;
        const float* src = x + e;
        v8bf p;
        #pragma unroll
        for (int i = 0; i < 8; ++i) p[i] = (bf16)src[i];
        *(v8bf*)(Xb + e) = p;
    } else {
        const int e = (blk - 4096) * 256 + threadIdx.x;   // [0, 3*512*512)
        const int which = e >> 18;
        const int rem = e & 262143;
        const int k = rem >> 9, n = rem & 511;
        const float* W = (which == 0) ? Wq : (which == 1) ? Wk : Wv;
        Wt[((size_t)which << 18) + ((size_t)n << 9) + k] = (bf16)W[rem];
    }
}

// ---------------------------------------------------------------------------
// Kernel 1: QKV projection, LDS-free.  Reads bf16 Xb (A-frags, contiguous) and
// transposed bf16 Wt (B-frags, contiguous); both L2-resident.
// Q output is pre-scaled by (1/sqrt(512))*log2(e) so attention runs in the
// exp2 domain.  V stored transposed Vt[b][d][n].
// Block: 256 threads (8 waves), tile 128(M) x 64(N); wave = 16-row strip.
// ---------------------------------------------------------------------------
__global__ __launch_bounds__(256) void qkv_gemm(
    const bf16* __restrict__ Xb, const bf16* __restrict__ Wt,
    const float* __restrict__ bq, const float* __restrict__ bk,
    const float* __restrict__ bv,
    bf16* __restrict__ Qb, bf16* __restrict__ Kb, bf16* __restrict__ Vt)
{
    const int which = blockIdx.z;
    const bf16* __restrict__ W     = Wt + ((size_t)which << 18);
    const float* __restrict__ bias = (which == 0) ? bq : (which == 1) ? bk : bv;

    const int row0 = blockIdx.x * 128;
    const int n0   = blockIdx.y * 64;
    const int t    = threadIdx.x;
    const int wave = t >> 5;
    const int lane = t & 31;
    const int lrow = lane & 15;
    const int hi   = lane >> 4;

    const bf16* xrow = Xb + (size_t)(row0 + wave * 16 + lrow) * D_;

    v8f acc[4];
    #pragma unroll
    for (int nt = 0; nt < 4; ++nt)
        #pragma unroll
        for (int i = 0; i < 8; ++i) acc[nt][i] = 0.f;

    #pragma unroll 4
    for (int k0 = 0; k0 < D_; k0 += 32) {
        v16bf a = cat16(ld8(xrow + k0 + hi * 8), ld8(xrow + k0 + 16 + hi * 8));
        #pragma unroll
        for (int nt = 0; nt < 4; ++nt) {
            const bf16* brow = W + (size_t)(n0 + nt * 16 + lrow) * D_ + k0 + hi * 8;
            v16bf bm = cat16(ld8(brow), ld8(brow + 16));
            acc[nt] = __builtin_amdgcn_wmma_f32_16x16x32_bf16(
                false, a, false, bm, (short)0, acc[nt], false, false);
        }
    }

    // epilogue: bias (+ Q pre-scale), bf16 stores
    const float qs   = 0.04419417382415922f * 1.4426950408889634f; // scale*log2e
    const float mulf = (which == 0) ? qs : 1.f;
    const int r0 = row0 + wave * 16;
    #pragma unroll
    for (int nt = 0; nt < 4; ++nt) {
        const int dcol = n0 + nt * 16 + lrow;
        const float bv_ = bias[dcol];
        if (which < 2) {
            bf16* outp = (which == 0) ? Qb : Kb;
            #pragma unroll
            for (int i = 0; i < 8; ++i) {
                const int M = hi * 8 + i;
                outp[(size_t)(r0 + M) * D_ + dcol] = (bf16)((acc[nt][i] + bv_) * mulf);
            }
        } else {
            const int b  = r0 / N_;
            const int nb = (r0 % N_) + hi * 8;
            v8bf pack;
            #pragma unroll
            for (int i = 0; i < 8; ++i) pack[i] = (bf16)(acc[nt][i] + bv_);
            *(v8bf*)&Vt[((size_t)b * D_ + dcol) * N_ + nb] = pack;
        }
    }
}

// ---------------------------------------------------------------------------
// Kernel 2: flash attention, 64 Q rows per block, 8 waves.
//   Q held in registers (16 persistent A-frags per wave, no LDS re-reads).
//   scores: wave w computes the full 16x16 tile (mt=w>>1, nt=w&1), 16 WMMA.
//   P@V:    wave w owns O tile-row mt and D-half w&1, 16 WMMA, 128-VGPR accum.
// K / V^T double-buffered in LDS via global_load_async_to_lds_b128; softmax is
// fully parallel (64 rows x 4 lanes, shfl_xor) in the exp2 domain.
// ---------------------------------------------------------------------------
// dynamic LDS layout (bytes):
//   [0      , 65536 )  KLs[2][32][512]  bf16
//   [65536  , 131072)  VLs[2][512][32]  bf16
//   [131072 , 139264)  Sls[64][32]      f32
//   [139264 , 143360)  Pls[64][32]      bf16
//   [143360 , 144128)  mls/lls/cls[64]  f32
#define ATTN_LDS_BYTES 144128

__global__ __launch_bounds__(256) void attn(
    const bf16* __restrict__ Qb, const bf16* __restrict__ Kb,
    const bf16* __restrict__ Vt, float* __restrict__ out)
{
    extern __shared__ char smem[];
    bf16 (*KLs)[32][512] = (bf16 (*)[32][512])(smem);
    bf16 (*VLs)[512][32] = (bf16 (*)[512][32])(smem + 65536);
    float (*Sls)[32]     = (float (*)[32])(smem + 131072);
    bf16  (*Pls)[32]     = (bf16  (*)[32])(smem + 139264);
    float* mls           = (float*)(smem + 143360);
    float* lls           = mls + 64;
    float* cls           = lls + 64;

    const int b    = blockIdx.y;
    const int m0   = blockIdx.x * 64;
    const int t    = threadIdx.x;
    const int wave = t >> 5;
    const int lane = t & 31;
    const int lrow = lane & 15;
    const int hi   = lane >> 4;
    const int mt   = wave >> 1;      // score/O m-tile (0..3)
    const int nt   = wave & 1;       // score n-tile
    const int hf   = wave & 1;       // O d-half (0..1)

    const bf16* Kbase = Kb + (size_t)b * N_ * D_;
    const bf16* Vbase = Vt + (size_t)b * D_ * N_;

    auto issue_kv = [&](int buf, int kb) {
        // K block: rows kb..kb+32, all of D -> contiguous 32 KB
        const bf16* gk = Kbase + (size_t)kb * D_;
        unsigned lk = (unsigned)(uintptr_t)&KLs[buf][0][0];
        #pragma unroll
        for (int j = 0; j < 8; ++j) {
            const unsigned off = (unsigned)(t + 256 * j) * 16u;
            async_copy16(lk + off, (const bf16*)((const char*)gk + off));
        }
        // V^T slice: 512 rows of 64 B, global row stride N*2 bytes
        unsigned lv = (unsigned)(uintptr_t)&VLs[buf][0][0];
        #pragma unroll
        for (int j = 0; j < 8; ++j) {
            const int idx = t + 256 * j;          // [0, 2048)
            const int d = idx >> 2, ch = idx & 3;
            async_copy16(lv + (unsigned)(d * 64 + ch * 16),
                         Vbase + (size_t)d * N_ + kb + ch * 8);
        }
    };

    issue_kv(0, 0);   // prologue prefetch

    if (t < 64) { mls[t] = -3.0e38f; lls[t] = 0.f; }

    // Q m-tile held in registers for the whole kernel (pre-scaled in qkv_gemm)
    v16bf qf[16];
    {
        const bf16* qrow = Qb + ((size_t)b * N_ + m0 + mt * 16 + lrow) * D_;
        #pragma unroll
        for (int kk = 0; kk < 16; ++kk)
            qf[kk] = cat16(ld8(qrow + kk * 32 + hi * 8),
                           ld8(qrow + kk * 32 + 16 + hi * 8));
    }

    v8f o[16];
    #pragma unroll
    for (int dt = 0; dt < 16; ++dt)
        #pragma unroll
        for (int i = 0; i < 8; ++i) o[dt][i] = 0.f;

    for (int it = 0; it < N_ / 32; ++it) {
        const int buf = it & 1;
        wait_async0();                 // current KV buffer landed (this wave)
        __syncthreads();               // publish staged data; close prev-iter reads

        if (it + 1 < N_ / 32) issue_kv(buf ^ 1, (it + 1) * 32);   // prefetch next

        // full 16x16 score tile for (mt, nt) over D=512
        v8f s;
        #pragma unroll
        for (int i = 0; i < 8; ++i) s[i] = 0.f;
        #pragma unroll
        for (int kk = 0; kk < 16; ++kk) {
            const bf16* brow = &KLs[buf][nt * 16 + lrow][kk * 32 + hi * 8];
            v16bf bm = cat16(ld8(brow), ld8(brow + 16));
            s = __builtin_amdgcn_wmma_f32_16x16x32_bf16(
                false, qf[kk], false, bm, (short)0, s, false, false);
        }
        #pragma unroll
        for (int i = 0; i < 8; ++i)
            Sls[mt * 16 + hi * 8 + i][nt * 16 + lrow] = s[i];
        __syncthreads();

        // parallel online softmax (exp2 domain): 64 rows x 4 lanes/row
        {
            const int r  = t >> 2;
            const int c0 = (t & 3) * 8;
            const float* srow = &Sls[r][c0];
            float vals[8];
            float mx = -3.0e38f;
            #pragma unroll
            for (int j = 0; j < 8; ++j) { vals[j] = srow[j]; mx = fmaxf(mx, vals[j]); }
            mx = fmaxf(mx, __shfl_xor(mx, 1, 32));
            mx = fmaxf(mx, __shfl_xor(mx, 2, 32));
            const float mold = mls[r];
            const float mnew = fmaxf(mold, mx);
            const float corr = exp2f(mold - mnew);
            float sum = 0.f;
            v8bf p8;
            #pragma unroll
            for (int j = 0; j < 8; ++j) {
                const float p = exp2f(vals[j] - mnew);
                sum += p;
                p8[j] = (bf16)p;
            }
            *(v8bf*)&Pls[r][c0] = p8;
            sum += __shfl_xor(sum, 1, 32);
            sum += __shfl_xor(sum, 2, 32);
            if ((t & 3) == 0) {
                lls[r] = lls[r] * corr + sum;
                mls[r] = mnew;
                cls[r] = corr;
            }
        }
        __syncthreads();

        // rescale O by correction factors, then O += P @ V
        float cv[8];
        #pragma unroll
        for (int i = 0; i < 8; ++i) cv[i] = cls[mt * 16 + hi * 8 + i];
        #pragma unroll
        for (int dt = 0; dt < 16; ++dt)
            #pragma unroll
            for (int i = 0; i < 8; ++i) o[dt][i] *= cv[i];

        const bf16* prow = &Pls[mt * 16 + lrow][0];
        v16bf pf = cat16(ld8(prow + hi * 8), ld8(prow + 16 + hi * 8));
        #pragma unroll
        for (int dt = 0; dt < 16; ++dt) {
            const bf16* vrow = &VLs[buf][hf * 256 + dt * 16 + lrow][hi * 8];
            v16bf vf = cat16(ld8(vrow), ld8(vrow + 16));
            o[dt] = __builtin_amdgcn_wmma_f32_16x16x32_bf16(
                false, pf, false, vf, (short)0, o[dt], false, false);
        }
    }

    // finalize: divide by softmax denominator, store fp32
    float lv[8];
    #pragma unroll
    for (int i = 0; i < 8; ++i) lv[i] = 1.f / lls[mt * 16 + hi * 8 + i];
    #pragma unroll
    for (int dt = 0; dt < 16; ++dt) {
        const int dcol = hf * 256 + dt * 16 + lrow;
        #pragma unroll
        for (int i = 0; i < 8; ++i) {
            const int M = mt * 16 + hi * 8 + i;
            out[((size_t)b * N_ + m0 + M) * D_ + dcol] = o[dt][i] * lv[i];
        }
    }
}

// ---------------------------------------------------------------------------
extern "C" void kernel_launch(void* const* d_in, const int* in_sizes, int n_in,
                              void* d_out, int out_size, void* d_ws, size_t ws_size,
                              hipStream_t stream)
{
    const float* x  = (const float*)d_in[0];
    const float* Wq = (const float*)d_in[1];
    const float* bq = (const float*)d_in[2];
    const float* Wk = (const float*)d_in[3];
    const float* bk = (const float*)d_in[4];
    const float* Wv = (const float*)d_in[5];
    const float* bv = (const float*)d_in[6];
    float* out = (float*)d_out;

    // workspace (bf16): Q | K | V^T | Xb | Wt  = 16+16+16+16+1.5 MB
    const size_t SZ = (size_t)B_ * N_ * D_;   // 8388608 elements
    bf16* Qb = (bf16*)d_ws;
    bf16* Kb = Qb + SZ;
    bf16* Vt = Kb + SZ;
    bf16* Xb = Vt + SZ;
    bf16* Wt = Xb + SZ;

    prep<<<dim3(7168), dim3(256), 0, stream>>>(x, Wq, Wk, Wv, Xb, Wt);

    dim3 g1(B_ * N_ / 128, D_ / 64, 3);
    qkv_gemm<<<g1, dim3(256), 0, stream>>>(Xb, Wt, bq, bk, bv, Qb, Kb, Vt);

    dim3 g2(N_ / 64, B_);
    attn<<<g2, dim3(256), ATTN_LDS_BYTES, stream>>>(Qb, Kb, Vt, out);
}
